// GMNLayerX_28432683499987
// MI455X (gfx1250) — matrix operand order repaired and checked by reading, not compiled, not run on hardware
//
#include <hip/hip_runtime.h>

typedef __attribute__((ext_vector_type(16))) _Float16 v16h;
typedef __attribute__((ext_vector_type(8)))  float    v8f;

#define HID  64
#define FIN  64
#define KE   160   // 2*FIN + 16 = 144, padded to multiple of 32
#define KN   192   // HID + FIN + HID
#define WPB  8     // waves per block (256 threads); ~94-105KB LDS -> 3 blocks per 320KB WGP

#define AS3 __attribute__((address_space(3)))

// ---- CDNA5 async global->LDS copy (ASYNCcnt-tracked, 16B per lane) ----
static __device__ __forceinline__ void async_copy16(void* lds, const void* g) {
  unsigned loff = (unsigned)(unsigned long long)(AS3 char*)lds;
  asm volatile("global_load_async_to_lds_b128 %0, %1, off"
               :: "v"(loff), "v"(g) : "memory");
}
static __device__ __forceinline__ void wait_async() {
  asm volatile("s_wait_asynccnt 0x0" ::: "memory");
}

// ---- WMMA fragment helpers (CDNA5 16x16x32 f16 layouts, wave32) ----
// A (16x32 f16): lane m = lane&15; lanes<16 hold K=0..15, lanes>=16 K=16..31
static __device__ __forceinline__ v16h lds_load_a(const _Float16* rowBase, int kc, int lh) {
  const int base = kc * 32 + lh * 16;
  v16h a;
#pragma unroll
  for (int i = 0; i < 16; ++i) a[i] = rowBase[base + i];
  return a;
}
// B (32x16 f16): lane n = lane&15; lanes<16 hold K=0..15, lanes>=16 K=16..31 (weights N-major)
static __device__ __forceinline__ v16h lds_load_b(const _Float16* colBase, int kc, int lh) {
  const int base = kc * 32 + lh * 16;
  v16h b;
#pragma unroll
  for (int i = 0; i < 16; ++i) b[i] = colBase[base + i];
  return b;
}
static __device__ __forceinline__ v8f wmma_f16(v16h a, v16h b, v8f c) {
  return __builtin_amdgcn_wmma_f32_16x16x32_f16(false, a, false, b, (short)0, c, false, false);
}

// ------------------------------------------------------------------
__global__ void zero_ws_kernel(float* __restrict__ p, long n) {
  long i = (long)blockIdx.x * blockDim.x + threadIdx.x;
  long stride = (long)gridDim.x * blockDim.x;
  for (; i < n; i += stride) p[i] = 0.0f;
}

// Pre-pack h -> f16 and weights -> f16 transposed (N-major) layouts.
__global__ void pack_kernel(const float* __restrict__ h,
                            const float* __restrict__ We1, const float* __restrict__ We2,
                            const float* __restrict__ Wc1, const float* __restrict__ Wn1,
                            const float* __restrict__ Wn2, const float* __restrict__ Wv1,
                            _Float16* __restrict__ h16,
                            _Float16* __restrict__ wt1, _Float16* __restrict__ wt2,
                            _Float16* __restrict__ wtc1, _Float16* __restrict__ wn1t,
                            _Float16* __restrict__ wn2t, _Float16* __restrict__ wv1t,
                            int N) {
  long i = (long)blockIdx.x * blockDim.x + threadIdx.x;
  long stride = (long)gridDim.x * blockDim.x;
  for (long j = i; j < (long)N * FIN; j += stride) h16[j] = (_Float16)h[j];
  for (long j = i; j < HID * KE; j += stride) {
    int n = (int)(j / KE), k = (int)(j % KE);
    wt1[j] = (k < 2 * FIN + 16) ? (_Float16)We1[k * HID + n] : (_Float16)0.0f;
  }
  for (long j = i; j < HID * KN; j += stride) {
    int n = (int)(j / KN), k = (int)(j % KN);
    wn1t[j] = (_Float16)Wn1[k * HID + n];
  }
  for (long j = i; j < HID * HID; j += stride) {
    int n = (int)(j / HID), k = (int)(j % HID);
    wt2[j]  = (_Float16)We2[k * HID + n];
    wtc1[j] = (_Float16)Wc1[k * HID + n];
    wn2t[j] = (_Float16)Wn2[k * FIN + n];
    wv1t[j] = (_Float16)Wv1[k * HID + n];
  }
}

// ------------------------------------------------------------------
// Edge kernel: one 16-edge tile per wave, 8 waves per block.
// LDS aliasing: `ein` (160 cols) is dead after GEMM1; its first 64 columns are
// reused as the edge_feat staging buffer (what was `mid2`). A __syncthreads()
// separates the last donor read from the first aliased write.
__global__ __launch_bounds__(WPB * 32)
void egnn_edge_kernel(const _Float16* __restrict__ h16, const float* __restrict__ Z,
                      const int* __restrict__ row, const int* __restrict__ col,
                      const _Float16* __restrict__ wt1g, const float* __restrict__ be1,
                      const _Float16* __restrict__ wt2g, const float* __restrict__ be2,
                      const _Float16* __restrict__ wtc1g, const float* __restrict__ bc1,
                      const float* __restrict__ Wc2,
                      float* __restrict__ seg, float* __restrict__ cntRow,
                      float* __restrict__ cntCol, float* __restrict__ agg,
                      float* __restrict__ others, int E) {
  __shared__ _Float16 Wt1[HID * KE];      // We1^T (K-padded), f16
  __shared__ _Float16 Wt2[HID * HID];     // We2^T
  __shared__ _Float16 Wtc1[HID * HID];    // Wc1^T
  __shared__ float sWc2[HID], sbe1[HID], sbe2[HID], sbc1[HID];
  __shared__ _Float16 ein[WPB][16][KE];   // e_in tile; later reused as edge_feat staging
  __shared__ _Float16 mid[WPB][16][HID];
  __shared__ int rIdx[WPB][16], cIdx[WPB][16];

  const int tid = threadIdx.x;
  // ---- async-copy packed f16 weights into LDS ----
  for (int i = tid; i < (HID * KE) / 8; i += WPB * 32)
    async_copy16(&Wt1[i * 8], wt1g + i * 8);
  for (int i = tid; i < (HID * HID) / 8; i += WPB * 32) {
    async_copy16(&Wt2[i * 8], wt2g + i * 8);
    async_copy16(&Wtc1[i * 8], wtc1g + i * 8);
  }
  for (int i = tid; i < HID; i += WPB * 32) {
    sWc2[i] = Wc2[i]; sbe1[i] = be1[i]; sbe2[i] = be2[i]; sbc1[i] = bc1[i];
  }

  const int wave = tid >> 5, lane = tid & 31;
  const int lh = lane >> 4, ln = lane & 15;
  const int eBase = (blockIdx.x * WPB + wave) * 16;

  // ---- phase 1: coord2radial (registers + ds) + async h16 gathers ----
  float dreg[12];
#pragma unroll
  for (int j = 0; j < 12; ++j) dreg[j] = 0.0f;

  if (lane < 16) {
    int m = lane;
    int e = eBase + m;
    bool valid = e < E;
    int ec = valid ? e : 0;
    int r = row[ec], c = col[ec];
    rIdx[wave][m] = valid ? r : -1;
    cIdx[wave][m] = valid ? c : -1;
#pragma unroll
    for (int j = 0; j < 12; ++j) dreg[j] = Z[(size_t)r * 12 + j] - Z[(size_t)c * 12 + j];
    float g[16]; float ss = 0.0f;
#pragma unroll
    for (int a = 0; a < 4; ++a)
#pragma unroll
      for (int b = 0; b < 4; ++b) {
        float v = dreg[a*3+0]*dreg[b*3+0] + dreg[a*3+1]*dreg[b*3+1] + dreg[a*3+2]*dreg[b*3+2];
        g[a*4+b] = v; ss += v * v;
      }
    float inv = 1.0f / fmaxf(sqrtf(ss), 1e-12f);
#pragma unroll
    for (int j = 0; j < 16; ++j) ein[wave][m][128 + j] = (_Float16)(g[j] * inv);
#pragma unroll
    for (int j = 2 * FIN + 16; j < KE; ++j) ein[wave][m][j] = (_Float16)0.0f;
  }
  {  // h16[row] / h16[col] gathers straight into LDS: 2 lanes per edge, 8 x B128 each
    int m = lane >> 1, half = lane & 1;
    int e = eBase + m;
    int ec = (e < E) ? e : 0;
    int node = half ? col[ec] : row[ec];
    const _Float16* src = h16 + (size_t)node * FIN;
#pragma unroll
    for (int j = 0; j < 8; ++j)
      async_copy16(&ein[wave][m][half * 64 + j * 8], src + j * 8);
  }
  wait_async();
  __syncthreads();

  // ---- GEMM1: relu(e_in @ We1 + be1), K=160 ----
  v8f acc[4];
#pragma unroll
  for (int nt = 0; nt < 4; ++nt) {
    float b0 = sbe1[nt * 16 + ln];
#pragma unroll
    for (int q = 0; q < 8; ++q) acc[nt][q] = b0;
  }
#pragma unroll
  for (int kc = 0; kc < KE / 32; ++kc) {
    v16h a = lds_load_a(&ein[wave][ln][0], kc, lh);
#pragma unroll
    for (int nt = 0; nt < 4; ++nt) {
      v16h b = lds_load_b(&Wt1[(nt * 16 + ln) * KE], kc, lh);
      acc[nt] = wmma_f16(a, b, acc[nt]);
    }
  }
#pragma unroll
  for (int nt = 0; nt < 4; ++nt)
#pragma unroll
    for (int q = 0; q < 8; ++q)
      mid[wave][q + lh * 8][nt * 16 + ln] = (_Float16)fmaxf(acc[nt][q], 0.0f);
  __syncthreads();   // last read of ein was in GEMM1 -> safe to alias below

  // ---- GEMM2: edge_feat = relu(mid @ We2 + be2), K=64; stage into ein[..][0..63] ----
  v8f ef[4];
#pragma unroll
  for (int nt = 0; nt < 4; ++nt) {
    float b0 = sbe2[nt * 16 + ln];
#pragma unroll
    for (int q = 0; q < 8; ++q) ef[nt][q] = b0;
  }
#pragma unroll
  for (int kc = 0; kc < 2; ++kc) {
    v16h a = lds_load_a(&mid[wave][ln][0], kc, lh);
#pragma unroll
    for (int nt = 0; nt < 4; ++nt) {
      v16h b = lds_load_b(&Wt2[(nt * 16 + ln) * HID], kc, lh);
      ef[nt] = wmma_f16(a, b, ef[nt]);
    }
  }
#pragma unroll
  for (int nt = 0; nt < 4; ++nt)
#pragma unroll
    for (int q = 0; q < 8; ++q) {
      float v = fmaxf(ef[nt][q], 0.0f);
      ef[nt][q] = v;
      ein[wave][q + lh * 8][nt * 16 + ln] = (_Float16)v;   // aliased mid2
    }
  __syncthreads();

  // ---- GEMM3: hid = relu(edge_feat @ Wc1 + bc1), K=64 ----
  v8f acc3[4];
#pragma unroll
  for (int nt = 0; nt < 4; ++nt) {
    float b0 = sbc1[nt * 16 + ln];
#pragma unroll
    for (int q = 0; q < 8; ++q) acc3[nt][q] = b0;
  }
#pragma unroll
  for (int kc = 0; kc < 2; ++kc) {
    v16h a = lds_load_a(&ein[wave][ln][0], kc, lh);        // aliased mid2
#pragma unroll
    for (int nt = 0; nt < 4; ++nt) {
      v16h b = lds_load_b(&Wtc1[(nt * 16 + ln) * HID], kc, lh);
      acc3[nt] = wmma_f16(a, b, acc3[nt]);
    }
  }
#pragma unroll
  for (int nt = 0; nt < 4; ++nt)
#pragma unroll
    for (int q = 0; q < 8; ++q)
      mid[wave][q + lh * 8][nt * 16 + ln] = (_Float16)fmaxf(acc3[nt][q], 0.0f);
  __syncthreads();

  // ---- phi = hid @ Wc2, trans scatter + counts (lane m owns edge m) ----
  if (lane < 16) {
    int m = lane;
    int rI = rIdx[wave][m], cI = cIdx[wave][m];
    if (rI >= 0) {
      float phi = 0.0f;
#pragma unroll
      for (int n = 0; n < HID; ++n) phi += (float)mid[wave][m][n] * sWc2[n];
#pragma unroll
      for (int j = 0; j < 12; ++j) atomicAdd(&seg[(size_t)rI * 12 + j], dreg[j] * phi);
      atomicAdd(&cntRow[rI], 1.0f);
      atomicAdd(&cntCol[cI], 1.0f);
    }
  }
  // ---- edge_feat scatter: agg (by row) + others (by col) ----
#pragma unroll
  for (int nt = 0; nt < 4; ++nt)
#pragma unroll
    for (int q = 0; q < 8; ++q) {
      int m = q + lh * 8;
      int rI = rIdx[wave][m], cI = cIdx[wave][m];
      if (rI >= 0) {
        float v = ef[nt][q];
        atomicAdd(&agg[(size_t)rI * HID + nt * 16 + ln], v);
        atomicAdd(&others[(size_t)cI * HID + nt * 16 + ln], v);
      }
    }
}

// ------------------------------------------------------------------
// Node kernel: one 16-node tile per wave, 8 waves per block.
// LDS aliasing: `nin` (192 cols) is dead after GEMM1; its first 64 columns are
// reused as the h_new staging buffer (what was `midB`).
__global__ __launch_bounds__(WPB * 32)
void egnn_node_kernel(const float* __restrict__ h, const _Float16* __restrict__ h16,
                      const float* __restrict__ Z,
                      const _Float16* __restrict__ wn1tg, const float* __restrict__ bn1,
                      const _Float16* __restrict__ wn2tg, const float* __restrict__ bn2,
                      const _Float16* __restrict__ wv1tg, const float* __restrict__ bv1,
                      const float* __restrict__ Wv2, const float* __restrict__ bv2,
                      const float* __restrict__ seg, const float* __restrict__ cntRow,
                      const float* __restrict__ cntCol, const float* __restrict__ agg,
                      const float* __restrict__ others,
                      float* __restrict__ out_h, float* __restrict__ out_Z,
                      float* __restrict__ out_x, float* __restrict__ out_v, int N) {
  __shared__ _Float16 Wn1t[HID * KN];
  __shared__ _Float16 Wn2t[FIN * HID];
  __shared__ _Float16 Wv1t[HID * FIN];
  __shared__ float sWv2[HID], sbn1[HID], sbn2[FIN], sbv1[HID];
  __shared__ _Float16 nin[WPB][16][KN];   // n_in tile; later reused as h_new staging
  __shared__ _Float16 midA[WPB][16][HID];
  __shared__ int nIdx[WPB][16];

  const int tid = threadIdx.x;
  for (int i = tid; i < (HID * KN) / 8; i += WPB * 32)
    async_copy16(&Wn1t[i * 8], wn1tg + i * 8);
  for (int i = tid; i < (HID * HID) / 8; i += WPB * 32) {
    async_copy16(&Wn2t[i * 8], wn2tg + i * 8);
    async_copy16(&Wv1t[i * 8], wv1tg + i * 8);
  }
  for (int i = tid; i < HID; i += WPB * 32) {
    sWv2[i] = Wv2[i]; sbn1[i] = bn1[i]; sbn2[i] = bn2[i]; sbv1[i] = bv1[i];
  }

  const int wave = tid >> 5, lane = tid & 31;
  const int lh = lane >> 4, ln = lane & 15;
  const int nBase = (blockIdx.x * WPB + wave) * 16;

  // ---- build n_in = [others/cnt_col | h | agg] ----
  {
    int m = ln;
    int node = nBase + m;
    bool valid = node < N;
    int nc = valid ? node : 0;
    if (lh == 0) {
      nIdx[wave][m] = valid ? node : -1;
      float ccnt = fmaxf(cntCol[nc], 1.0f);
#pragma unroll 4
      for (int j = 0; j < 64; ++j)
        nin[wave][m][j] = (_Float16)(others[(size_t)nc * HID + j] / ccnt);
    } else {
#pragma unroll 4
      for (int j = 0; j < 64; ++j)
        nin[wave][m][128 + j] = (_Float16)agg[(size_t)nc * HID + j];
    }
  }
  {  // h section straight from packed f16: 2 lanes per node, 4 x B128 each
    int m = lane >> 1, half = lane & 1;
    int node = nBase + m;
    int nc = (node < N) ? node : 0;
    const _Float16* src = h16 + (size_t)nc * FIN + half * 32;
#pragma unroll
    for (int j = 0; j < 4; ++j)
      async_copy16(&nin[wave][m][64 + half * 32 + j * 8], src + j * 8);
  }
  wait_async();
  __syncthreads();

  // ---- GEMM1: relu(n_in @ Wn1 + bn1), K=192 ----
  v8f acc[4];
#pragma unroll
  for (int nt = 0; nt < 4; ++nt) {
    float b0 = sbn1[nt * 16 + ln];
#pragma unroll
    for (int q = 0; q < 8; ++q) acc[nt][q] = b0;
  }
#pragma unroll
  for (int kc = 0; kc < KN / 32; ++kc) {
    v16h a = lds_load_a(&nin[wave][ln][0], kc, lh);
#pragma unroll
    for (int nt = 0; nt < 4; ++nt) {
      v16h b = lds_load_b(&Wn1t[(nt * 16 + ln) * KN], kc, lh);
      acc[nt] = wmma_f16(a, b, acc[nt]);
    }
  }
#pragma unroll
  for (int nt = 0; nt < 4; ++nt)
#pragma unroll
    for (int q = 0; q < 8; ++q)
      midA[wave][q + lh * 8][nt * 16 + ln] = (_Float16)fmaxf(acc[nt][q], 0.0f);
  __syncthreads();   // last read of nin was in GEMM1 -> safe to alias below

  // ---- GEMM2: h_new = h + (midA @ Wn2 + bn2), K=64; stage into nin[..][0..63] ----
  v8f acc2[4];
#pragma unroll
  for (int nt = 0; nt < 4; ++nt) {
    float b0 = sbn2[nt * 16 + ln];
#pragma unroll
    for (int q = 0; q < 8; ++q) acc2[nt][q] = b0;
  }
#pragma unroll
  for (int kc = 0; kc < 2; ++kc) {
    v16h a = lds_load_a(&midA[wave][ln][0], kc, lh);
#pragma unroll
    for (int nt = 0; nt < 4; ++nt) {
      v16h b = lds_load_b(&Wn2t[(nt * 16 + ln) * HID], kc, lh);
      acc2[nt] = wmma_f16(a, b, acc2[nt]);
    }
  }
#pragma unroll
  for (int nt = 0; nt < 4; ++nt)
#pragma unroll
    for (int q = 0; q < 8; ++q) {
      int m = q + lh * 8;
      int node = nIdx[wave][m];
      int nc = (node >= 0) ? node : 0;
      int nco = nt * 16 + ln;
      float hn = acc2[nt][q] + h[(size_t)nc * FIN + nco];   // f32 residual add
      if (node >= 0) out_h[(size_t)node * FIN + nco] = hn;
      nin[wave][m][nco] = (_Float16)hn;                     // aliased midB
    }
  __syncthreads();

  // ---- GEMM3: relu(h_new @ Wv1 + bv1), K=64 ----
  v8f acc3[4];
#pragma unroll
  for (int nt = 0; nt < 4; ++nt) {
    float b0 = sbv1[nt * 16 + ln];
#pragma unroll
    for (int q = 0; q < 8; ++q) acc3[nt][q] = b0;
  }
#pragma unroll
  for (int kc = 0; kc < 2; ++kc) {
    v16h a = lds_load_a(&nin[wave][ln][0], kc, lh);         // aliased midB
#pragma unroll
    for (int nt = 0; nt < 4; ++nt) {
      v16h b = lds_load_b(&Wv1t[(nt * 16 + ln) * HID], kc, lh);
      acc3[nt] = wmma_f16(a, b, acc3[nt]);
    }
  }
#pragma unroll
  for (int nt = 0; nt < 4; ++nt)
#pragma unroll
    for (int q = 0; q < 8; ++q)
      midA[wave][q + lh * 8][nt * 16 + ln] = (_Float16)fmaxf(acc3[nt][q], 0.0f);
  __syncthreads();

  // ---- scale = midA @ Wv2 + bv2; coordinate finish (lane m owns node m) ----
  if (lane < 16) {
    int m = lane;
    int node = nIdx[wave][m];
    if (node >= 0) {
      float sc = bv2[0];
#pragma unroll
      for (int n = 0; n < HID; ++n) sc += (float)midA[wave][m][n] * sWv2[n];
      float crow = fmaxf(cntRow[node], 1.0f);
      float f[12];
#pragma unroll
      for (int j = 0; j < 12; ++j) {
        f[j] = seg[(size_t)node * 12 + j] / crow;
        out_Z[(size_t)node * 12 + j] = Z[(size_t)node * 12 + j] + f[j];
      }
#pragma unroll
      for (int t = 0; t < 3; ++t) {
        float vn = sc * Z[(size_t)node * 12 + 3 + t] + f[t];
        out_v[(size_t)node * 3 + t] = vn;
        out_x[(size_t)node * 3 + t] = Z[(size_t)node * 12 + t] + vn;
      }
    }
  }
}

// ------------------------------------------------------------------
extern "C" void kernel_launch(void* const* d_in, const int* in_sizes, int n_in,
                              void* d_out, int out_size, void* d_ws, size_t ws_size,
                              hipStream_t stream) {
  (void)n_in; (void)out_size; (void)ws_size;
  const float* h   = (const float*)d_in[0];
  const float* Z   = (const float*)d_in[1];
  const int*   row = (const int*)d_in[2];
  const int*   col = (const int*)d_in[3];
  const float* We1 = (const float*)d_in[4];
  const float* be1 = (const float*)d_in[5];
  const float* We2 = (const float*)d_in[6];
  const float* be2 = (const float*)d_in[7];
  const float* Wn1 = (const float*)d_in[8];
  const float* bn1 = (const float*)d_in[9];
  const float* Wn2 = (const float*)d_in[10];
  const float* bn2 = (const float*)d_in[11];
  const float* Wc1 = (const float*)d_in[12];
  const float* bc1 = (const float*)d_in[13];
  const float* Wc2 = (const float*)d_in[14];
  const float* Wv1 = (const float*)d_in[15];
  const float* bv1 = (const float*)d_in[16];
  const float* Wv2 = (const float*)d_in[17];
  const float* bv2 = (const float*)d_in[18];

  const int E = in_sizes[2];
  const int N = in_sizes[0] / FIN;

  // workspace layout (64B-aligned sections):
  // f32: seg[N*12] | cntRow[N] | cntCol[N] | agg[N*64] | others[N*64]
  // f16: h16[N*64] | wt1[64*160] | wt2[64*64] | wtc1[64*64] | wn1t[64*192] | wn2t[64*64] | wv1t[64*64]
  char* wsb = (char*)d_ws;
  float* seg    = (float*)wsb;
  float* cntRow = seg + (size_t)N * 12;
  float* cntCol = cntRow + N;
  float* agg    = cntCol + N;
  float* others = agg + (size_t)N * HID;
  const long wsElems = (long)N * (12 + 1 + 1 + HID + HID);
  size_t off = ((size_t)wsElems * 4 + 63) & ~(size_t)63;
  _Float16* h16   = (_Float16*)(wsb + off);
  _Float16* wt1g  = h16 + (size_t)N * FIN;
  _Float16* wt2g  = wt1g + HID * KE;
  _Float16* wtc1g = wt2g + HID * HID;
  _Float16* wn1tg = wtc1g + HID * HID;
  _Float16* wn2tg = wn1tg + HID * KN;
  _Float16* wv1tg = wn2tg + FIN * HID;

  // outputs: h_new[N*64] | Z_new[N*12] | x_new[N*3] | v_new[N*3]
  float* out   = (float*)d_out;
  float* out_h = out;
  float* out_Z = out_h + (size_t)N * FIN;
  float* out_x = out_Z + (size_t)N * 12;
  float* out_v = out_x + (size_t)N * 3;

  zero_ws_kernel<<<2048, 256, 0, stream>>>(seg, wsElems);
  pack_kernel<<<2048, 256, 0, stream>>>(h, We1, We2, Wc1, Wn1, Wn2, Wv1,
                                        h16, wt1g, wt2g, wtc1g, wn1tg, wn2tg, wv1tg, N);

  const int eTiles = (E + 15) / 16;
  const int eBlocks = (eTiles + WPB - 1) / WPB;
  egnn_edge_kernel<<<eBlocks, WPB * 32, 0, stream>>>(
      h16, Z, row, col, wt1g, be1, wt2g, be2, wtc1g, bc1, Wc2,
      seg, cntRow, cntCol, agg, others, E);

  const int nTiles = (N + 15) / 16;
  const int nBlocks = (nTiles + WPB - 1) / WPB;
  egnn_node_kernel<<<nBlocks, WPB * 32, 0, stream>>>(
      h, h16, Z, wn1tg, bn1, wn2tg, bn2, wv1tg, bv1, Wv2, bv2,
      seg, cntRow, cntCol, agg, others,
      out_h, out_Z, out_x, out_v, N);
}
